// GC_T_15281493639232
// MI455X (gfx1250) — compile-verified, matrix operand order checked by reading, not run on hardware
//
#include <hip/hip_runtime.h>
#include <math.h>

#define NN 256
#define BB 16
#define FF 128
#define HH 8
#define DH 16
#define LL 9
#define BNODES (BB*NN)   // 4096
#define EE 65536

typedef __attribute__((ext_vector_type(16))) _Float16 v16h;
typedef __attribute__((ext_vector_type(8)))  float    v8f;

__device__ __forceinline__ v8f wmma_f32_f16(v16h a, v16h b, v8f c) {
  // D = A(16x32 f16) * B(32x16 f16) + C(16x16 f32)
  return __builtin_amdgcn_wmma_f32_16x16x32_f16(false, a, false, b, (short)0, c, false, false);
}

__device__ __forceinline__ float wave_red_sum(float v) {
  #pragma unroll
  for (int m = 16; m >= 1; m >>= 1) v += __shfl_xor(v, m, 32);
  return v;
}
__device__ __forceinline__ float wave_red_max(float v) {
  #pragma unroll
  for (int m = 16; m >= 1; m >>= 1) v = fmaxf(v, __shfl_xor(v, m, 32));
  return v;
}

// K index base for f16 A/B fragments (ISA 16-bit A-matrix 16x32 layout)
__device__ __forceinline__ int frag_kbase(int j, int half) {
  return (j < 4) ? (8*half + 2*j) : (16 + 8*half + 2*(j-4));
}

// async 16B-per-lane copy: global f16 -> LDS, tracked by ASYNCcnt
__device__ __forceinline__ void async_copy_b128(const _Float16* g, _Float16* l) {
  unsigned lds_off = (unsigned)(uintptr_t)l;   // low 32 bits of LDS generic addr = LDS byte offset
  asm volatile("global_load_async_to_lds_b128 %0, %1, off"
               :: "v"(lds_off), "v"(g) : "memory");
}
__device__ __forceinline__ void async_wait(int n) {
  if (n == 0) asm volatile("s_wait_asynccnt 0x0" ::: "memory");
  else        asm volatile("s_wait_asynccnt 0x2" ::: "memory");
}

// ---------------------------------------------------------------------------
// Generic GEMM on f16 operands: Y/Y16 = act( alpha * X[M,K] @ W[Nc,K]^T + bias )
// block = 256 threads (8 waves), tile 64x64, K multiple of 32.
// Double-buffered async global->LDS staging. act: 0 none, 1 relu, 2 gelu.
// ---------------------------------------------------------------------------
__global__ __launch_bounds__(256) void gemm_xwt(
    const _Float16* __restrict__ X, const _Float16* __restrict__ W,
    float* __restrict__ Y, _Float16* __restrict__ Y16,
    int M, int Nc, int K, float alpha, const float* __restrict__ bias, int act)
{
  __shared__ _Float16 Xs[2][64][40];
  __shared__ _Float16 Ws[2][64][40];
  const int tid  = threadIdx.x;
  const int lane = tid & 31;
  const int wid  = tid >> 5;
  const int half = lane >> 4;
  const int lr   = lane & 15;
  const int rt   = wid & 3;       // row tile 0..3
  const int ch   = wid >> 2;      // column half 0..1 (2 col tiles each)
  const int row0 = blockIdx.x * 64;
  const int col0 = blockIdx.y * 64;
  // async chunk mapping: 256 threads x 16B cover one 64x32-f16 tile
  const int arow = tid >> 2;           // 0..63
  const int acol = (tid & 3) * 8;      // 0,8,16,24 (halfs)

  const int nk = K >> 5;
  // prologue: stage 0
  async_copy_b128(X + (size_t)(row0 + arow) * K + acol, &Xs[0][arow][acol]);
  async_copy_b128(W + (size_t)(col0 + arow) * K + acol, &Ws[0][arow][acol]);

  v8f c0 = {}; v8f c1 = {};
  for (int kc = 0; kc < nk; ++kc) {
    const int cur = kc & 1;
    if (kc + 1 < nk) {
      const int k0 = (kc + 1) << 5;
      async_copy_b128(X + (size_t)(row0 + arow) * K + k0 + acol, &Xs[cur ^ 1][arow][acol]);
      async_copy_b128(W + (size_t)(col0 + arow) * K + k0 + acol, &Ws[cur ^ 1][arow][acol]);
      async_wait(2);
    } else {
      async_wait(0);
    }
    __syncthreads();
    v16h a;
    #pragma unroll
    for (int j = 0; j < 8; ++j) {
      int kb = frag_kbase(j, half);
      a[2*j]   = Xs[cur][rt*16 + lr][kb];
      a[2*j+1] = Xs[cur][rt*16 + lr][kb+1];
    }
    #pragma unroll
    for (int t = 0; t < 2; ++t) {
      int ct = ch*2 + t;
      v16h bf;
      #pragma unroll
      for (int j = 0; j < 8; ++j) {
        int kb = frag_kbase(j, half);
        bf[2*j]   = Ws[cur][ct*16 + lr][kb];
        bf[2*j+1] = Ws[cur][ct*16 + lr][kb+1];
      }
      if (t == 0) c0 = wmma_f32_f16(a, bf, c0);
      else        c1 = wmma_f32_f16(a, bf, c1);
    }
    __syncthreads();
  }
  #pragma unroll
  for (int t = 0; t < 2; ++t) {
    v8f c = t ? c1 : c0;
    int col = col0 + (ch*2 + t)*16 + lr;
    float bv = bias ? bias[col] : 0.0f;
    #pragma unroll
    for (int j = 0; j < 8; ++j) {
      int row = row0 + rt*16 + j + 8*half;
      float v = alpha * c[j] + bv;
      if (act == 1)      v = fmaxf(v, 0.0f);
      else if (act == 2) v = 0.5f * v * (1.0f + erff(v * 0.70710678118f));
      if (Y)   Y[(size_t)row * Nc + col]   = v;
      if (Y16) Y16[(size_t)row * Nc + col] = (_Float16)v;
    }
  }
}

// ---------------------------------------------------------------------------
// Batched MCL GEMM: Out[b] = A16[b] @ B16[b], all 256x256 (f16 in, f32 out)
// grid (4,4,B); async double-buffered staging.
// ---------------------------------------------------------------------------
__global__ __launch_bounds__(256) void gemm_mcl(
    const _Float16* __restrict__ A16, const _Float16* __restrict__ B16,
    float* __restrict__ Out)
{
  __shared__ _Float16 As[2][64][40];
  __shared__ _Float16 Bs[2][32][72];
  const int tid = threadIdx.x, lane = tid & 31, wid = tid >> 5;
  const int half = lane >> 4, lr = lane & 15;
  const int rt = wid & 3, ch = wid >> 2;
  const int b = blockIdx.z;
  const int row0 = blockIdx.x * 64, col0 = blockIdx.y * 64;
  const _Float16* A = A16 + (size_t)b * NN * NN;
  const _Float16* Bm = B16 + (size_t)b * NN * NN;
  float* O = Out + (size_t)b * NN * NN;
  // A tile 64x32: chunk mapping as in gemm_xwt
  const int arow = tid >> 2, acol = (tid & 3) * 8;
  // B tile 32x64: 32 rows x 64 halfs
  const int brow = tid >> 3, bcol = (tid & 7) * 8;

  async_copy_b128(A + (size_t)(row0 + arow) * NN + acol,        &As[0][arow][acol]);
  async_copy_b128(Bm + (size_t)brow * NN + col0 + bcol,         &Bs[0][brow][bcol]);

  v8f c0 = {}; v8f c1 = {};
  const int nk = NN >> 5;
  for (int kc = 0; kc < nk; ++kc) {
    const int cur = kc & 1;
    if (kc + 1 < nk) {
      const int k0 = (kc + 1) << 5;
      async_copy_b128(A + (size_t)(row0 + arow) * NN + k0 + acol, &As[cur ^ 1][arow][acol]);
      async_copy_b128(Bm + (size_t)(k0 + brow) * NN + col0 + bcol, &Bs[cur ^ 1][brow][bcol]);
      async_wait(2);
    } else {
      async_wait(0);
    }
    __syncthreads();
    v16h a;
    #pragma unroll
    for (int j = 0; j < 8; ++j) {
      int kb = frag_kbase(j, half);
      a[2*j]   = As[cur][rt*16 + lr][kb];
      a[2*j+1] = As[cur][rt*16 + lr][kb+1];
    }
    #pragma unroll
    for (int t = 0; t < 2; ++t) {
      int ct = ch*2 + t;
      v16h bf;
      #pragma unroll
      for (int j = 0; j < 8; ++j) {
        int kb = frag_kbase(j, half);
        bf[2*j]   = Bs[cur][kb][ct*16 + lr];
        bf[2*j+1] = Bs[cur][kb+1][ct*16 + lr];
      }
      if (t == 0) c0 = wmma_f32_f16(a, bf, c0);
      else        c1 = wmma_f32_f16(a, bf, c1);
    }
    __syncthreads();
  }
  #pragma unroll
  for (int t = 0; t < 2; ++t) {
    v8f c = t ? c1 : c0;
    int col = col0 + (ch*2 + t)*16 + lr;
    #pragma unroll
    for (int j = 0; j < 8; ++j)
      O[(size_t)(row0 + rt*16 + j + 8*half) * NN + col] = c[j];
  }
}

// ---------------------------------------------------------------------------
// MCL row normalization (in-place safe), inflate => v^6 first; writes f16 copy.
// ---------------------------------------------------------------------------
__global__ __launch_bounds__(256) void mcl_rownorm(const float* In, float* Out,
                                                   _Float16* Out16, int inflate)
{
  __shared__ float part[8];
  int row = blockIdx.x;
  int tid = threadIdx.x;
  float v = In[(size_t)row * NN + tid];
  if (inflate) { float v2 = v*v; v = v2*v2*v2; }
  float s = wave_red_sum(v);
  if ((tid & 31) == 0) part[tid >> 5] = s;
  __syncthreads();
  float tot = part[0]+part[1]+part[2]+part[3]+part[4]+part[5]+part[6]+part[7];
  float r = v / (tot + 1e-6f);
  Out[(size_t)row * NN + tid] = r;
  Out16[(size_t)row * NN + tid] = (_Float16)r;
}

// f32 -> f16 conversion
__global__ void cvt_f16(const float* __restrict__ src, _Float16* __restrict__ dst, int n)
{
  int i = blockIdx.x * 256 + threadIdx.x;
  if (i < n) dst[i] = (_Float16)src[i];
}

// ---------------------------------------------------------------------------
// adjacency builders
// ---------------------------------------------------------------------------
__global__ void adj_edges(const int* __restrict__ erow, const int* __restrict__ ecol,
                          float* __restrict__ m)
{
  int e = blockIdx.x * 256 + threadIdx.x;
  if (e >= EE) return;
  int r = erow[e], c = ecol[e];
  int b = r / NN, es = r % NN, et = c % NN;
  m[((size_t)b * NN + es) * NN + et] = 1.0f;
}
__global__ void adj_diag(const int* __restrict__ n_counts, float* __restrict__ m)
{
  int i = blockIdx.x * 256 + threadIdx.x;
  int b = i / NN, n = i % NN;
  if (n < n_counts[b]) m[((size_t)b * NN + n) * NN + n] = 1.0f;
}

// ---------------------------------------------------------------------------
// degree
// ---------------------------------------------------------------------------
__global__ void deg_scatter(const int* __restrict__ erow, float* __restrict__ degr)
{
  int e = blockIdx.x * 256 + threadIdx.x;
  if (e < EE) atomicAdd(&degr[erow[e]], 1.0f);
}
__global__ void deg_fin(float* deg, float* dis)
{
  int i = blockIdx.x * 256 + threadIdx.x;
  float d = deg[i] + 1.0f;
  deg[i] = d;
  dis[i] = rsqrtf(d);
}

// ---------------------------------------------------------------------------
// GCN edge messages: msg = dis[r]*dis[c]*relu(h[r]+edge_attr@We^T) -> atomic agg[c]
// ---------------------------------------------------------------------------
__global__ __launch_bounds__(256) void edge_msg_kernel(
    const float* __restrict__ edge_attr, const float* __restrict__ Wedge,
    const int* __restrict__ erow, const int* __restrict__ ecol,
    const float* __restrict__ h, const float* __restrict__ dis,
    float* __restrict__ agg)
{
  __shared__ float Wl[FF * 7];
  for (int i = threadIdx.x; i < FF * 7; i += 256) Wl[i] = Wedge[i];
  __syncthreads();
  int e = blockIdx.x * 2 + (threadIdx.x >> 7);
  if (e >= EE) return;
  int f = threadIdx.x & 127;
  int r = erow[e], c = ecol[e];
  float en = dis[r] * dis[c];
  const float* ea = edge_attr + (size_t)e * 7;
  float ee = 0.0f;
  #pragma unroll
  for (int j = 0; j < 7; ++j) ee += Wl[f*7 + j] * ea[j];
  float msg = en * fmaxf(h[(size_t)r * FF + f] + ee, 0.0f);
  atomicAdd(&agg[(size_t)c * FF + f], msg);
}

// g = agg + relu(h + root_emb[root]) / deg
__global__ void gcn_combine(const float* __restrict__ agg, const float* __restrict__ h,
                            const float* __restrict__ remb, const int* __restrict__ root,
                            const float* __restrict__ deg, float* __restrict__ g)
{
  int i = blockIdx.x * 256 + threadIdx.x;   // over BNODES*FF
  int n = i >> 7, f = i & 127;
  float e = remb[root[n] * FF + f];
  g[i] = agg[i] + fmaxf(h[i] + e, 0.0f) / deg[n];
}

// ---------------------------------------------------------------------------
// LayerNorm over F=128; one wave per row. Optional f16 shadow output.
//   mode 0: Y = LN(X + (Res?Res:0))*g + b
//   mode 1: Y = relu(LN(X)*g + b) + Res
// (pointers may alias -> no restrict)
// ---------------------------------------------------------------------------
__global__ __launch_bounds__(256) void ln_kernel(
    const float* X, const float* Res, float* Y, _Float16* Y16,
    const float* g, const float* bta, int M, int mode)
{
  int wid = threadIdx.x >> 5, lane = threadIdx.x & 31;
  int row = blockIdx.x * 8 + wid;
  if (row >= M) return;
  const float* xp = X + (size_t)row * FF;
  const float* rp = Res ? Res + (size_t)row * FF : nullptr;
  float vals[4];
  #pragma unroll
  for (int t = 0; t < 4; ++t) {
    int f = lane + 32*t;
    vals[t] = xp[f];
    if (mode == 0 && rp) vals[t] += rp[f];
  }
  float mu = wave_red_sum(vals[0]+vals[1]+vals[2]+vals[3]) * (1.0f/FF);
  float var = 0.0f;
  #pragma unroll
  for (int t = 0; t < 4; ++t) { float d = vals[t]-mu; var += d*d; }
  var = wave_red_sum(var) * (1.0f/FF);
  float inv = rsqrtf(var + 1e-5f);
  float* yp = Y + (size_t)row * FF;
  #pragma unroll
  for (int t = 0; t < 4; ++t) {
    int f = lane + 32*t;
    float y = (vals[t]-mu)*inv*g[f] + bta[f];
    if (mode == 1) y = fmaxf(y, 0.0f) + rp[f];
    yp[f] = y;
    if (Y16) Y16[(size_t)row * FF + f] = (_Float16)y;
  }
}

// ---------------------------------------------------------------------------
// Fused biased attention per (32 rows, head, batch). grid (8, H, B), 256 thr.
// q/k/v are f16; S = q@k^T (+pad/mcl bias), softmax, O = P@v (f16 out).
// ---------------------------------------------------------------------------
__global__ __launch_bounds__(256) void attn_kernel(
    const _Float16* __restrict__ q, const _Float16* __restrict__ k,
    const _Float16* __restrict__ v,
    const float* __restrict__ mcl, const float* __restrict__ mclw,
    const int* __restrict__ n_counts, _Float16* __restrict__ o)
{
  __shared__ _Float16 qs[32][16];
  __shared__ _Float16 ks[256][16];
  __shared__ _Float16 vs[256][16];
  __shared__ float    S[32][260];
  const int tid = threadIdx.x, lane = tid & 31, wid = tid >> 5;
  const int half = lane >> 4, lr = lane & 15;
  const int rb = blockIdx.x, h = blockIdx.y, b = blockIdx.z;
  const int n0 = rb * 32;
  const int cnt = n_counts[b];
  const float mw = mclw[h];

  #pragma unroll
  for (int e = 0; e < 2; ++e) {
    int i = tid*2 + e; int r = i >> 4, d = i & 15;
    qs[r][d] = q[(size_t)(b*NN + n0 + r)*FF + h*DH + d];
  }
  #pragma unroll
  for (int e = 0; e < 16; ++e) {
    int i = tid*16 + e; int r = i >> 4, d = i & 15;
    ks[r][d] = k[(size_t)(b*NN + r)*FF + h*DH + d];
    vs[r][d] = v[(size_t)(b*NN + r)*FF + h*DH + d];
  }
  __syncthreads();

  // ---- scores: wave wid -> row tile (wid>>2), col tiles (wid&3)*4 .. +3 ----
  const int rt = wid >> 2;
  v16h a;
  #pragma unroll
  for (int j = 0; j < 8; ++j) {
    int kb = frag_kbase(j, half);
    a[2*j]   = (kb   < 16) ? qs[rt*16 + lr][kb]   : (_Float16)0.0f;
    a[2*j+1] = (kb+1 < 16) ? qs[rt*16 + lr][kb+1] : (_Float16)0.0f;
  }
  #pragma unroll
  for (int t = 0; t < 4; ++t) {
    int ct = (wid & 3) * 4 + t;
    v16h bf;
    #pragma unroll
    for (int j = 0; j < 8; ++j) {
      int kb = frag_kbase(j, half);
      bf[2*j]   = (kb   < 16) ? ks[ct*16 + lr][kb]   : (_Float16)0.0f;
      bf[2*j+1] = (kb+1 < 16) ? ks[ct*16 + lr][kb+1] : (_Float16)0.0f;
    }
    v8f c = {};
    c = wmma_f32_f16(a, bf, c);
    #pragma unroll
    for (int j = 0; j < 8; ++j) {
      int rl = rt*16 + j + 8*half;
      int ng = n0 + rl;
      int mg = ct*16 + lr;
      float pb = ((ng < cnt) && (mg < cnt)) ? 0.0f : -1024.0f;
      float mv = mcl[((size_t)(b*NN) + ng) * NN + mg];
      S[rl][mg] = c[j] + pb + mv * mw;
    }
  }
  __syncthreads();

  // ---- softmax: wave wid handles rows wid*4 .. wid*4+3 ----
  #pragma unroll
  for (int rr = 0; rr < 4; ++rr) {
    int rl = wid*4 + rr;
    float vv[8];
    float mx = -1e30f;
    #pragma unroll
    for (int t = 0; t < 8; ++t) { vv[t] = S[rl][lane + 32*t]; mx = fmaxf(mx, vv[t]); }
    mx = wave_red_max(mx);
    float sum = 0.0f;
    #pragma unroll
    for (int t = 0; t < 8; ++t) { vv[t] = expf(vv[t] - mx); sum += vv[t]; }
    sum = wave_red_sum(sum);
    float inv = 1.0f / sum;
    #pragma unroll
    for (int t = 0; t < 8; ++t) S[rl][lane + 32*t] = vv[t] * inv;
  }
  __syncthreads();

  // ---- O = P @ v : waves 0,1 (one 16-row tile each), K=256 in 8 chunks ----
  if (wid < 2) {
    v8f c = {};
    for (int kc = 0; kc < 8; ++kc) {
      v16h pa, pb;
      #pragma unroll
      for (int j = 0; j < 8; ++j) {
        int kb = frag_kbase(j, half);
        pa[2*j]   = (_Float16)S[wid*16 + lr][kc*32 + kb];
        pa[2*j+1] = (_Float16)S[wid*16 + lr][kc*32 + kb + 1];
        pb[2*j]   = vs[kc*32 + kb][lr];
        pb[2*j+1] = vs[kc*32 + kb + 1][lr];
      }
      c = wmma_f32_f16(pa, pb, c);
    }
    #pragma unroll
    for (int j = 0; j < 8; ++j) {
      int rl = wid*16 + j + 8*half;
      o[(size_t)(b*NN + n0 + rl)*FF + h*DH + lr] = (_Float16)c[j];
    }
  }
}

// ===========================================================================
extern "C" void kernel_launch(void* const* d_in, const int* in_sizes, int n_in,
                              void* d_out, int out_size, void* d_ws, size_t ws_size,
                              hipStream_t stream)
{
  const float* input_x      = (const float*)d_in[0];
  const float* edge_attr    = (const float*)d_in[1];
  const float* gcn_lin_w    = (const float*)d_in[2];
  const float* gcn_root_emb = (const float*)d_in[3];
  const float* gcn_edge_w   = (const float*)d_in[4];
  const float* gcn_norm_g   = (const float*)d_in[5];
  const float* gcn_norm_b   = (const float*)d_in[6];
  const float* ln_in_g      = (const float*)d_in[7];
  const float* ln_in_b      = (const float*)d_in[8];
  const float* ln_ffn_g     = (const float*)d_in[9];
  const float* ln_ffn_b     = (const float*)d_in[10];
  const float* mcl_w        = (const float*)d_in[11];
  const float* q_w          = (const float*)d_in[12];
  const float* k_w          = (const float*)d_in[13];
  const float* v_w          = (const float*)d_in[14];
  const float* merge_w      = (const float*)d_in[15];
  const float* merge_b      = (const float*)d_in[16];
  const float* ffn_w1       = (const float*)d_in[17];
  const float* ffn_b1       = (const float*)d_in[18];
  const float* ffn_w2       = (const float*)d_in[19];
  const float* ffn_b2       = (const float*)d_in[20];
  const float* final_g      = (const float*)d_in[21];
  const float* final_b      = (const float*)d_in[22];
  const int*   edge_index   = (const int*)d_in[23];
  const int*   n_counts     = (const int*)d_in[24];
  const int*   root         = (const int*)d_in[25];
  const int* erow = edge_index;
  const int* ecol = edge_index + EE;

  // ---- workspace bump allocator (256-byte aligned regions) ----
  char* base = (char*)d_ws;
  size_t woff = 0;
  auto alloc = [&](size_t bytes) -> void* {
    void* p = base + woff;
    woff += (bytes + 255) & ~(size_t)255;
    return p;
  };
  const size_t MATSZ = (size_t)BB * NN * NN;   // 1,048,576 elems
  const size_t ACT   = (size_t)BNODES * FF;    //   524,288 elems

  float* m_buf = (float*)alloc(MATSZ * 4);
  float* rA    = (float*)alloc(MATSZ * 4);
  float* rB    = (float*)alloc(MATSZ * 4);
  float* deg   = (float*)alloc(BNODES * 4);
  float* dis   = (float*)alloc(BNODES * 4);
  float* out   = (float*)alloc(ACT * 4);
  float* hbuf  = (float*)alloc(ACT * 4);
  float* aggb  = (float*)alloc(ACT * 4);
  float* gbuf  = (float*)alloc(ACT * 4);
  float* tmp   = (float*)alloc(ACT * 4);
  float* xbuf  = (float*)alloc(ACT * 4);
  float* fbuf  = (float*)alloc(ACT * 4);
  // f16 shadows
  _Float16* m16    = (_Float16*)alloc(MATSZ * 2);
  _Float16* rA16   = (_Float16*)alloc(MATSZ * 2);
  _Float16* rB16   = (_Float16*)alloc(MATSZ * 2);
  _Float16* out16  = (_Float16*)alloc(ACT * 2);
  _Float16* xbuf16 = (_Float16*)alloc(ACT * 2);
  _Float16* hid16  = (_Float16*)alloc(2 * ACT * 2);
  _Float16* ob16   = (_Float16*)alloc(ACT * 2);
  _Float16* q16    = (_Float16*)alloc(ACT * 2);
  _Float16* k16    = (_Float16*)alloc(ACT * 2);
  _Float16* v16    = (_Float16*)alloc(ACT * 2);
  // f16 weights (all layers)
  const size_t WSQ = (size_t)LL * FF * FF;       // 147456
  const size_t WFN = (size_t)LL * 2 * FF * FF;   // 294912
  _Float16* lin16 = (_Float16*)alloc(WSQ * 2);
  _Float16* qw16  = (_Float16*)alloc(WSQ * 2);
  _Float16* kw16  = (_Float16*)alloc(WSQ * 2);
  _Float16* vw16  = (_Float16*)alloc(WSQ * 2);
  _Float16* mw16  = (_Float16*)alloc(WSQ * 2);
  _Float16* f116  = (_Float16*)alloc(WFN * 2);
  _Float16* f216  = (_Float16*)alloc(WFN * 2);

  // ---- weight conversion (once) ----
  cvt_f16<<<(int)((WSQ + 255) / 256), 256, 0, stream>>>(gcn_lin_w, lin16, (int)WSQ);
  cvt_f16<<<(int)((WSQ + 255) / 256), 256, 0, stream>>>(q_w, qw16, (int)WSQ);
  cvt_f16<<<(int)((WSQ + 255) / 256), 256, 0, stream>>>(k_w, kw16, (int)WSQ);
  cvt_f16<<<(int)((WSQ + 255) / 256), 256, 0, stream>>>(v_w, vw16, (int)WSQ);
  cvt_f16<<<(int)((WSQ + 255) / 256), 256, 0, stream>>>(merge_w, mw16, (int)WSQ);
  cvt_f16<<<(int)((WFN + 255) / 256), 256, 0, stream>>>(ffn_w1, f116, (int)WFN);
  cvt_f16<<<(int)((WFN + 255) / 256), 256, 0, stream>>>(ffn_w2, f216, (int)WFN);

  // ---- adjacency + MCL ----
  hipMemsetAsync(m_buf, 0, MATSZ * 4, stream);
  hipMemsetAsync(deg, 0, BNODES * 4, stream);
  adj_edges<<<EE/256, 256, 0, stream>>>(erow, ecol, m_buf);
  adj_diag<<<BNODES/256, 256, 0, stream>>>(n_counts, m_buf);
  mcl_rownorm<<<BNODES, 256, 0, stream>>>(m_buf, m_buf, m16, 0);
  _Float16* src16 = m16;
  float*    dst   = rB;  _Float16* dst16 = rB16;
  for (int it = 0; it < 6; ++it) {
    gemm_mcl<<<dim3(4, 4, BB), 256, 0, stream>>>(m16, src16, dst);
    mcl_rownorm<<<BNODES, 256, 0, stream>>>(dst, dst, dst16, ((it + 1) % 3 == 0) ? 1 : 0);
    src16 = dst16;
    if (dst == rB) { dst = rA; dst16 = rA16; }
    else           { dst = rB; dst16 = rB16; }
  }
  const float* mcl = rA;   // 6 iterations end in rA

  // ---- degree normalization ----
  deg_scatter<<<EE/256, 256, 0, stream>>>(erow, deg);
  deg_fin<<<BNODES/256, 256, 0, stream>>>(deg, dis);

  // ---- out = input_x (f32 + f16 shadow) ----
  hipMemcpyAsync(out, input_x, ACT * 4, hipMemcpyDeviceToDevice, stream);
  cvt_f16<<<(int)(ACT / 256), 256, 0, stream>>>(input_x, out16, (int)ACT);

  const float qscale = 0.25f;   // dh^-0.5, dh=16
  for (int l = 0; l < LL; ++l) {
    // GCN
    gemm_xwt<<<dim3(64, 2), 256, 0, stream>>>(out16, lin16 + (size_t)l*FF*FF,
                                              hbuf, nullptr, BNODES, FF, FF, 1.0f, nullptr, 0);
    hipMemsetAsync(aggb, 0, ACT * 4, stream);
    edge_msg_kernel<<<EE/2, 256, 0, stream>>>(edge_attr, gcn_edge_w + (size_t)l*FF*7,
                                              erow, ecol, hbuf, dis, aggb);
    gcn_combine<<<(BNODES*FF)/256, 256, 0, stream>>>(aggb, hbuf,
                                                     gcn_root_emb + (size_t)l*2*FF,
                                                     root, deg, gbuf);
    ln_kernel<<<BNODES/8, 256, 0, stream>>>(gbuf, out, out, out16,
                                            gcn_norm_g + l*FF, gcn_norm_b + l*FF,
                                            BNODES, 1);
    // Attention
    gemm_xwt<<<dim3(64, 2), 256, 0, stream>>>(out16, qw16 + (size_t)l*FF*FF,
                                              nullptr, q16, BNODES, FF, FF, qscale, nullptr, 0);
    gemm_xwt<<<dim3(64, 2), 256, 0, stream>>>(out16, kw16 + (size_t)l*FF*FF,
                                              nullptr, k16, BNODES, FF, FF, 1.0f, nullptr, 0);
    gemm_xwt<<<dim3(64, 2), 256, 0, stream>>>(out16, vw16 + (size_t)l*FF*FF,
                                              nullptr, v16, BNODES, FF, FF, 1.0f, nullptr, 0);
    attn_kernel<<<dim3(8, HH, BB), 256, 0, stream>>>(q16, k16, v16, mcl,
                                                     mcl_w + (size_t)l*HH, n_counts, ob16);
    gemm_xwt<<<dim3(64, 2), 256, 0, stream>>>(ob16, mw16 + (size_t)l*FF*FF,
                                              tmp, nullptr, BNODES, FF, FF, 1.0f,
                                              merge_b + l*FF, 0);
    ln_kernel<<<BNODES/8, 256, 0, stream>>>(tmp, out, xbuf, xbuf16,
                                            ln_in_g + l*FF, ln_in_b + l*FF, BNODES, 0);
    // FFN
    gemm_xwt<<<dim3(64, 4), 256, 0, stream>>>(xbuf16, f116 + (size_t)l*2*FF*FF,
                                              nullptr, hid16, BNODES, 2*FF, FF, 1.0f,
                                              ffn_b1 + l*2*FF, 2);
    gemm_xwt<<<dim3(64, 2), 256, 0, stream>>>(hid16, f216 + (size_t)l*FF*2*FF,
                                              fbuf, nullptr, BNODES, FF, 2*FF, 1.0f,
                                              ffn_b2 + l*FF, 0);
    ln_kernel<<<BNODES/8, 256, 0, stream>>>(fbuf, xbuf, out, out16,
                                            ln_ffn_g + l*FF, ln_ffn_b + l*FF, BNODES, 0);
  }

  // final layernorm -> d_out
  ln_kernel<<<BNODES/8, 256, 0, stream>>>(out, nullptr, (float*)d_out, nullptr,
                                          final_g, final_b, BNODES, 0);
}
